// EALSTM_63290638074629
// MI455X (gfx1250) — matrix operand order repaired and checked by reading, not compiled, not run on hardware
//
#include <hip/hip_runtime.h>
#include <hip/hip_bf16.h>
#include <stdint.h>

// Problem constants (from reference)
#define DYNC 32
#define STATC 27
#define HID 256
#define SEQ 1024
#define BATCH 128
#define POUT 24
#define G3 768          // 3*H gate columns
#define KTOT 288        // fused K = H + DYN
#define KT 9            // K tiles of 32
#define NWG 8           // workgroups in recurrent kernel
#define NTW 6           // N tiles (16 wide) owned per WG: 2 f + 2 o + 2 g
#define MT 8            // M tiles (128/16)

typedef __attribute__((ext_vector_type(16))) __bf16 v16bf;
typedef __attribute__((ext_vector_type(8)))  float  v8f;
typedef __attribute__((ext_vector_type(4)))  unsigned int v4u;
typedef __attribute__((ext_vector_type(8)))  int v8i;
typedef __attribute__((ext_vector_type(4)))  int v4i;

union Frag { v16bf v; v4u q[2]; };

// Workspace layout (bytes)
#define OFF_WFRAG   0u
#define WFRAG_USH   (KT*48*512)               // 221184 ushorts = 442368 B
#define OFF_XDSWZ   442368u
#define XDSWZ_USH   (SEQ*MT*512)              // 4194304 ushorts = 8 MB
#define OFF_HBUF    (442368u + 8388608u)      // 2 * 128*256 bf16 = 131072 B
#define OFF_IGATE   (OFF_HBUF + 131072u)      // 128*256 f32 = 131072 B
#define OFF_PROJ    (OFF_IGATE + 131072u)     // 128*1024 f32 = 524288 B
#define OFF_BAR     (OFF_PROJ + 524288u)      // barrier counter

static __device__ __forceinline__ unsigned short f2bf(float f) {
  union { float f; unsigned int u; } x; x.f = f;
  unsigned int u = x.u;
  unsigned int r = u + 0x7FFFu + ((u >> 16) & 1u);   // round-to-nearest-even
  return (unsigned short)(r >> 16);
}

__global__ void k_zero(unsigned int* p, int n) {
  int i = blockIdx.x * blockDim.x + threadIdx.x;
  if (i < n) p[i] = 0u;
}

// i_gate = sigmoid(x_s @ W_sh + b_s); x_s[b][s] = input_seq[0, b, 31+s] (ref's flat-reshape quirk)
__global__ void k_igate(const float* __restrict__ iseq, const float* __restrict__ wsh,
                        const float* __restrict__ bs, float* __restrict__ ig) {
  int b = blockIdx.x, j = threadIdx.x;
  float a = bs[j];
  for (int s = 0; s < STATC; ++s)
    a += iseq[b * 58 + 31 + s] * wsh[s * HID + j];
  ig[b * HID + j] = 1.f / (1.f + __expf(-a));
}

// Pre-swizzle [W_hh ; W_ih] into bf16 B-fragments per ISA layout:
// B 32x16: lane L elem e -> N = nt*16 + L%16, K = kt*32 + (e<8 ? 8*(L/16)+e : 16 + 8*(L/16) + e-8)
__global__ void k_wfrag(const float* __restrict__ whh, const float* __restrict__ wih,
                        unsigned short* __restrict__ wf) {
  int idx = blockIdx.x * blockDim.x + threadIdx.x;
  if (idx >= KT * 48 * 512) return;
  int e = idx & 15, L = (idx >> 4) & 31, fragIdx = idx >> 9;
  int jn = fragIdx % NTW, kt = (fragIdx / NTW) % KT, g = fragIdx / (NTW * KT);
  int nt = (jn < 2) ? (2 * g + jn) : (jn < 4) ? (16 + 2 * g + jn - 2) : (32 + 2 * g + jn - 4);
  int n = nt * 16 + (L & 15);
  int k = kt * 32 + ((e < 8) ? (8 * (L >> 4) + e) : (16 + 8 * (L >> 4) + (e - 8)));
  float v = (k < HID) ? whh[k * G3 + n] : wih[(k - HID) * G3 + n];
  wf[idx] = f2bf(v);
}

// Pre-swizzle x_d (concat dyn[:31], seq_y) into bf16 A-fragments per ISA layout:
// A 16x32: lane L elem e -> M = mt*16 + L%16, K = (e<8 ? 8*(L/16)+e : 16 + 8*(L/16) + e-8)
__global__ void k_xdswz(const float* __restrict__ iseq, const float* __restrict__ sy,
                        unsigned short* __restrict__ xd) {
  int idx = blockIdx.x * blockDim.x + threadIdx.x;
  int e = idx & 15, L = (idx >> 4) & 31;
  int mt = (idx >> 9) & 7, t = idx >> 12;
  int hlf = L >> 4;
  int m = mt * 16 + (L & 15);
  int d = (e < 8) ? (8 * hlf + e) : (16 + 8 * hlf + (e - 8));
  float v = (d < 31) ? iseq[(m * SEQ + t) * 58 + d] : sy[m * SEQ + t];
  xd[idx] = f2bf(v);
}

// Persistent fused EA-LSTM recurrence. 8 WGs (one WGP each), 256 threads = 8 waves.
// WG g owns hidden units [g*32, g*32+32) -> gate tiles {2g,2g+1, 16+2g,16+2g+1, 32+2g,32+2g+1}.
// Wave w owns batch rows [w*16, w*16+16).
__global__ void __launch_bounds__(256) k_lstm(
    const unsigned short* __restrict__ wfrag, const unsigned short* __restrict__ xdswz,
    const float* __restrict__ igate, const float* __restrict__ bias,
    const float* __restrict__ w1, unsigned short* __restrict__ hbuf,
    float* __restrict__ proj, unsigned int* __restrict__ bar) {
  __shared__ unsigned short smw[KT * NTW * 512];   // 54 KB weight slice; sole LDS object -> offset 0

  const int tid = threadIdx.x;
  const int g   = blockIdx.x;
  const int w   = tid >> 5;     // wave index == M tile
  const int L   = tid & 31;
  const int hlf = L >> 4;
  const int lq  = L & 15;

  // TDM fill of the LDS weight slice: one linear 54 KB tile, issued by wave 0 only
  // (tensor ops ignore EXEC; the tid<32 branch is wave-uniform so waves 1-7 skip it).
  if (tid < 32) {
    unsigned long long ga = (unsigned long long)(uintptr_t)(wfrag + (size_t)g * KT * NTW * 512);
    // D# group 0: count=1 | lds_addr | global_addr[56:0] | type=2
    v4u d0 = { 1u,
               0u,                                    // smw LDS byte offset
               (unsigned)(ga & 0xFFFFFFFFu),
               (unsigned)(((ga >> 32) & 0x1FFFFFFu) | 0x80000000u) };
    // D# group 1: data_size=3 (8B elems); tensor_dim0 = tile_dim0 = 6912 (54KB);
    // tensor_dim1 = tile_dim1 = 1; dim0_stride = 6912; workgroup_mask = 0.
    v8i d1 = { (int)0x00030000,
               (int)(6912u << 16),
               (int)(1u << 16),
               (int)(6912u << 16),
               1,
               6912,
               0, 0 };
    v4i dz = {0, 0, 0, 0};
#if __clang_major__ >= 23
    v8i dz8 = {0, 0, 0, 0, 0, 0, 0, 0};
    __builtin_amdgcn_tensor_load_to_lds(d0, d1, dz, dz, dz8, 0);
#else
    __builtin_amdgcn_tensor_load_to_lds(d0, d1, dz, dz, 0);
#endif
    __builtin_amdgcn_s_wait_tensorcnt(0);
  }
  // The TDM descriptor carries the LDS address as an integer, so the compiler
  // cannot see that smw was written. Escape smw's address into an asm memory
  // clobber so the B-fragment ds_loads below are not folded as undef.
  asm volatile("" : : "v"(smw) : "memory");
  __syncthreads();

  const int mrow = w * 16 + 8 * hlf;   // base C-row for this lane's accumulator regs
  const int aRow = w * 16 + lq;        // A-fragment row for this lane
  const int huB  = g * 32 + lq;        // hidden-unit base (j adds 16)

  float biasv[NTW];
#pragma unroll
  for (int jn = 0; jn < NTW; ++jn) {
    int nt = (jn < 2) ? (2 * g + jn) : (jn < 4) ? (16 + 2 * g + jn - 2) : (32 + 2 * g + jn - 4);
    biasv[jn] = bias[nt * 16 + lq];
  }
  float igv[2][8], cst[2][8], w1l[2];
#pragma unroll
  for (int j = 0; j < 2; ++j) {
    w1l[j] = w1[huB + j * 16];
#pragma unroll
    for (int r = 0; r < 8; ++r) {
      igv[j][r] = igate[(mrow + r) * HID + huB + j * 16];
      cst[j][r] = 0.f;
    }
  }

  for (int t = 0; t < SEQ; ++t) {
    const unsigned short* hb_cur = hbuf + (t & 1) * BATCH * HID;
    unsigned short* hb_nxt = hbuf + ((t & 1) ^ 1) * BATCH * HID;

    if (t + 1 < SEQ)
      __builtin_prefetch(xdswz + (((t + 1) * MT + w) * 32 + L) * 16, 0, 1);

    v8f acc[NTW];
#pragma unroll
    for (int jn = 0; jn < NTW; ++jn) {
      float b = biasv[jn];
      v8f v = {b, b, b, b, b, b, b, b};
      acc[jn] = v;                      // gates = bias + [h|x] @ [Whh;Wih]
    }

#pragma unroll
    for (int kt = 0; kt < KT; ++kt) {
      Frag a;
      if (kt < 8) {                     // h part of A, from ping-pong buffer
        const unsigned short* p = hb_cur + aRow * HID + kt * 32 + 8 * hlf;
        a.q[0] = *(const v4u*)(p);      // K = kt*32 + 8h + [0..7]
        a.q[1] = *(const v4u*)(p + 16); // K = kt*32 + 16 + 8h + [0..7]
      } else {                          // x_d part of A, pre-swizzled
        const unsigned short* p = xdswz + ((t * MT + w) * 32 + L) * 16;
        a.q[0] = *(const v4u*)(p);
        a.q[1] = *(const v4u*)(p + 8);
      }
#pragma unroll
      for (int jn = 0; jn < NTW; ++jn) {
        Frag bf;
        const unsigned short* bp = smw + (kt * NTW + jn) * 512 + L * 16;
        bf.q[0] = *(const v4u*)(bp);
        bf.q[1] = *(const v4u*)(bp + 8);
        acc[jn] = __builtin_amdgcn_wmma_f32_16x16x32_bf16(
            false, a.v, false, bf.v, (short)0, acc[jn], false, false);
      }
    }

    // Elementwise LSTM update + bf16 h write + on-the-fly w1 projection
#pragma unroll
    for (int r = 0; r < 8; ++r) {
      float pv = 0.f;
#pragma unroll
      for (int j = 0; j < 2; ++j) {
        float fg = acc[j][r], og = acc[2 + j][r], gg = acc[4 + j][r];
        float c = cst[j][r];
        float sf = 1.f / (1.f + __expf(-fg));
        float so = 1.f / (1.f + __expf(-og));
        float tg = 1.f - 2.f / (__expf(2.f * gg) + 1.f);
        c = sf * c + igv[j][r] * tg;
        float tc = 1.f - 2.f / (__expf(2.f * c) + 1.f);
        float h = so * tc;
        cst[j][r] = c;
        hb_nxt[(mrow + r) * HID + huB + j * 16] = f2bf(h);
        pv += h * w1l[j];
      }
      pv += __shfl_xor(pv, 1, 32);
      pv += __shfl_xor(pv, 2, 32);
      pv += __shfl_xor(pv, 4, 32);
      pv += __shfl_xor(pv, 8, 32);
      if (lq == 0) atomicAdd(&proj[(mrow + r) * SEQ + t], pv);
    }

    // Inter-workgroup barrier (8 co-resident WGs), release/acquire through L2
    __threadfence();
    __syncthreads();
    if (tid == 0) {
      __hip_atomic_fetch_add(bar, 1u, __ATOMIC_RELEASE, __HIP_MEMORY_SCOPE_AGENT);
      unsigned tgt = (unsigned)(NWG * (t + 1));
      while (__hip_atomic_load(bar, __ATOMIC_ACQUIRE, __HIP_MEMORY_SCOPE_AGENT) < tgt)
        __builtin_amdgcn_s_sleep(2);
    }
    __syncthreads();
    __threadfence();
  }
}

// out[b,p] = b2[p] + sum_s (proj[b,s] + b1) * w2[p,s]
__global__ void k_final(const float* __restrict__ proj, const float* __restrict__ b1,
                        const float* __restrict__ w2, const float* __restrict__ b2,
                        float* __restrict__ out) {
  int idx = blockIdx.x * blockDim.x + threadIdx.x;
  if (idx >= BATCH * POUT) return;
  int b = idx / POUT, p = idx % POUT;
  float b1v = b1[0];
  float a = b2[p];
  for (int s = 0; s < SEQ; ++s)
    a += (proj[b * SEQ + s] + b1v) * w2[p * SEQ + s];
  out[b * POUT + p] = a;
}

extern "C" void kernel_launch(void* const* d_in, const int* in_sizes, int n_in,
                              void* d_out, int out_size, void* d_ws, size_t ws_size,
                              hipStream_t stream) {
  (void)in_sizes; (void)n_in; (void)out_size; (void)ws_size;
  const float* iseq = (const float*)d_in[0];
  const float* sy   = (const float*)d_in[1];
  const float* wih  = (const float*)d_in[2];
  const float* whh  = (const float*)d_in[3];
  const float* wsh  = (const float*)d_in[4];
  const float* bias = (const float*)d_in[5];
  const float* bs   = (const float*)d_in[6];
  const float* w1   = (const float*)d_in[7];
  const float* b1   = (const float*)d_in[8];
  const float* w2   = (const float*)d_in[9];
  const float* b2   = (const float*)d_in[10];

  char* ws = (char*)d_ws;
  unsigned short* wfrag = (unsigned short*)(ws + OFF_WFRAG);
  unsigned short* xdswz = (unsigned short*)(ws + OFF_XDSWZ);
  unsigned short* hbuf  = (unsigned short*)(ws + OFF_HBUF);
  float* igate          = (float*)(ws + OFF_IGATE);
  float* proj           = (float*)(ws + OFF_PROJ);
  unsigned int* bar     = (unsigned int*)(ws + OFF_BAR);

  // Zero: h ping-pong buffer slot 0, proj accumulator, barrier counter (every call)
  k_zero<<<(BATCH * HID / 2 + 255) / 256, 256, 0, stream>>>((unsigned int*)hbuf, BATCH * HID / 2);
  k_zero<<<(BATCH * SEQ + 255) / 256, 256, 0, stream>>>((unsigned int*)proj, BATCH * SEQ);
  k_zero<<<1, 32, 0, stream>>>(bar, 1);

  k_igate<<<BATCH, HID, 0, stream>>>(iseq, wsh, bs, igate);
  k_wfrag<<<(KT * 48 * 512) / 256, 256, 0, stream>>>(whh, wih, wfrag);
  k_xdswz<<<(SEQ * MT * 512) / 256, 256, 0, stream>>>(iseq, sy, xdswz);

  k_lstm<<<NWG, 256, 0, stream>>>(wfrag, xdswz, igate, bias, w1, hbuf, proj, bar);

  k_final<<<(BATCH * POUT + 255) / 256, 256, 0, stream>>>(proj, b1, w2, b2, (float*)d_out);
}
